// RandLANetRes_32323923870347
// MI455X (gfx1250) — matrix operand order repaired and checked by reading, not compiled, not run on hardware
//
#include <hip/hip_runtime.h>
#include <hip/hip_bf16.h>

typedef _Float16 v8h  __attribute__((ext_vector_type(8)));
typedef _Float16 v16h __attribute__((ext_vector_type(16)));
typedef float    v8f  __attribute__((ext_vector_type(8)));

#define WMMA_F16(a,b,c) __builtin_amdgcn_wmma_f32_16x16x32_f16(false,(a),false,(b),(short)0,(c),false,false)

__device__ __forceinline__ v16h cat8(v8h lo, v8h hi){
  return __builtin_shufflevector(lo,hi,0,1,2,3,4,5,6,7,8,9,10,11,12,13,14,15);
}

// A-operand (16-bit, 16x32 chunk) from row-major LDS tile [16 x stride] halves.
// lane = c*16 + m ; a[i<8] = T[m][kc + c*8 + i], a[i>=8] = T[m][kc + 16 + c*8 + (i-8)]
__device__ __forceinline__ v16h ldA(const _Float16* t, int stride, int m, int c, int kc){
  const _Float16* p = t + m*stride + kc + c*8;
  v8h lo = *(const v8h*)(p);
  v8h hi = *(const v8h*)(p + 16);
  return cat8(lo, hi);
}

// B-operand (16-bit, 32x16 chunk) from K-major (transposed) weights Wt[n][Ktot].
// lane = c*16 + n' ; b[i<8] = W[kc + c*8 + i][n], b[i>=8] = W[kc + 16 + c*8 + (i-8)][n]
__device__ __forceinline__ v16h ldB(const _Float16* wt, int Ktot, int n, int c, int kc){
  const _Float16* p = wt + n*Ktot + kc + c*8;
  v8h lo = *(const v8h*)(p);
  v8h hi = *(const v8h*)(p + 16);
  return cat8(lo, hi);
}

// One RandLA conv layer. Block = 512 threads = 16 waves = 16 nodes (one node / wave,
// K=16 neighbors == one WMMA M-tile). D_E = 96 fixed.
template<int DIN, int DOUT, bool RESID>
__global__ __launch_bounds__(512)
void randla_conv(const float* __restrict__ xin,   // [N,DIN] input features
                 const float* __restrict__ pos,   // [N,3]
                 const int*   __restrict__ idx,   // [N,16]
                 const float* __restrict__ Wp, const float* __restrict__ bp,   // [10,32],[32]
                 const float* __restrict__ Wa, const float* __restrict__ ba,   // [96,96],[96]
                 const float* __restrict__ Wg, const float* __restrict__ bg,   // [96,DOUT],[DOUT]
                 const float* __restrict__ Ws, const float* __restrict__ bs,   // [64,DOUT],[DOUT]
                 const float* __restrict__ xres,                               // [N,64] (RESID)
                 float* __restrict__ out, int N)
{
  __shared__ __attribute__((aligned(16))) _Float16 sWpT[32*32];          // [n][k] (k padded 10->32)
  __shared__ __attribute__((aligned(16))) _Float16 sWaT[96*96];          // [n][k]
  __shared__ __attribute__((aligned(16))) _Float16 sWgT[DOUT*96];        // [n][k]
  __shared__ __attribute__((aligned(16))) _Float16 sWsT[RESID ? DOUT*64 : 8];
  __shared__ __attribute__((aligned(16))) _Float16 sXres[RESID ? 16*64 : 8];
  __shared__ float sbp[32], sba[96], sbg[DOUT];
  __shared__ float sbs[RESID ? DOUT : 8];
  __shared__ __attribute__((aligned(16))) _Float16 sFij[16][16*96];      // per-wave edge features
  __shared__ __attribute__((aligned(16))) _Float16 sRel[16][16*32];      // per-wave rel tile (K padded)
  __shared__ __attribute__((aligned(16))) _Float16 sAgg[16*96];          // block aggr tile (node-major)

  const int tid  = threadIdx.x;
  const int w    = tid >> 5;       // wave id == node slot in block
  const int lane = tid & 31;
  const int c    = lane >> 4;      // half-wave select
  const int np   = lane & 15;      // M (A) / N' (B,C) position
  const int node = blockIdx.x * 16 + w;

  // ---- stage weights (fp32 -> fp16, transposed to K-major) ----
  for (int i = tid; i < 32*32; i += 512){ int n=i>>5, k=i&31; sWpT[n*32+k] = (k<10) ? (_Float16)Wp[k*32+n] : (_Float16)0.f; }
  for (int i = tid; i < 96*96; i += 512){ int n=i/96, k=i%96; sWaT[n*96+k] = (_Float16)Wa[k*96+n]; }
  for (int i = tid; i < DOUT*96; i += 512){ int n=i/96, k=i%96; sWgT[n*96+k] = (_Float16)Wg[k*DOUT+n]; }
  if constexpr (RESID){
    for (int i = tid; i < DOUT*64; i += 512){ int n=i>>6, k=i&63; sWsT[n*64+k] = (_Float16)Ws[k*DOUT+n]; }
    for (int i = tid; i < 16*64; i += 512){ int r=i>>6, f=i&63; sXres[i] = (_Float16)xres[(blockIdx.x*16+r)*64 + f]; }
    for (int i = tid; i < DOUT; i += 512) sbs[i] = bs[i];
  }
  for (int i = tid; i < 32;   i += 512) sbp[i] = bp[i];
  for (int i = tid; i < 96;   i += 512) sba[i] = ba[i];
  for (int i = tid; i < DOUT; i += 512) sbg[i] = bg[i];
  __syncthreads();

  _Float16* fij = sFij[w];
  _Float16* rel = sRel[w];

  // ---- gather x[idx] -> fij cols [0,DIN) (gather rows are 256B lines; x fits in 192MB L2) ----
  for (int t = lane; t < 16*DIN; t += 32){
    int e = t / DIN, f = t % DIN;
    int j = idx[node*16 + e];
    fij[e*96 + f] = (_Float16)xin[j*DIN + f];
  }
  // ---- rel features (10, zero-padded to 32) : lanes 0..15, one edge each ----
  if (lane < 16){
    int e = lane;
    int j = idx[node*16 + e];
    float pix = pos[node*3+0], piy = pos[node*3+1], piz = pos[node*3+2];
    float pjx = pos[j*3+0],    pjy = pos[j*3+1],    pjz = pos[j*3+2];
    float vx = pix-pjx, vy = piy-pjy, vz = piz-pjz;
    float d  = sqrtf(vx*vx + vy*vy + vz*vz);
    float rf[10] = {pix,piy,piz,pjx,pjy,pjz,vx,vy,vz,d};
    #pragma unroll
    for (int k = 0; k < 10; ++k) rel[e*32+k] = (_Float16)rf[k];
    #pragma unroll
    for (int k = 10; k < 32; ++k) rel[e*32+k] = (_Float16)0.f;
  }
  __builtin_amdgcn_wave_barrier();   // same-wave LDS is in-order; keep the compiler honest

  // ---- rij = relu(rel @ Wp + bp) : M16 K32 N32 -> fij cols [64,96) ----
  {
    v16h a = ldA(rel, 32, np, c, 0);
    #pragma unroll
    for (int t = 0; t < 2; ++t){
      v8f acc = {};
      acc = WMMA_F16(a, ldB(sWpT, 32, t*16+np, c, 0), acc);
      int col = t*16 + np;
      float bias = sbp[col];
      #pragma unroll
      for (int r = 0; r < 8; ++r){
        float v = acc[r] + bias; v = v > 0.f ? v : 0.f;
        fij[(r + 8*c)*96 + 64 + col] = (_Float16)v;   // C layout: row = r + 8c, col fixed
      }
    }
  }
  __builtin_amdgcn_wave_barrier();

  // ---- g = relu(fij @ Wa + ba) : M16 K96 N96 (18 WMMAs) ----
  v8f gacc[6];
  {
    v16h a0 = ldA(fij, 96, np, c, 0);
    v16h a1 = ldA(fij, 96, np, c, 32);
    v16h a2 = ldA(fij, 96, np, c, 64);
    #pragma unroll
    for (int t = 0; t < 6; ++t){
      v8f acc = {};
      acc = WMMA_F16(a0, ldB(sWaT, 96, t*16+np, c, 0),  acc);
      acc = WMMA_F16(a1, ldB(sWaT, 96, t*16+np, c, 32), acc);
      acc = WMMA_F16(a2, ldB(sWaT, 96, t*16+np, c, 64), acc);
      float bias = sba[t*16+np];
      #pragma unroll
      for (int r = 0; r < 8; ++r){ float v = acc[r] + bias; acc[r] = v > 0.f ? v : 0.f; }
      gacc[t] = acc;
    }
  }

  // ---- softmax over the 96 features per edge, then sum_k s*fij -> aggr[96] ----
  // C-layout row m = r + 8c is constant across each 16-lane half => reduce with width-16 shfl.
  float partial[6];
  {
    float mx[8], inv[8];
    #pragma unroll
    for (int r = 0; r < 8; ++r){
      float m = gacc[0][r];
      #pragma unroll
      for (int t = 1; t < 6; ++t) m = fmaxf(m, gacc[t][r]);
      #pragma unroll
      for (int off = 1; off < 16; off <<= 1) m = fmaxf(m, __shfl_xor(m, off, 16));
      mx[r] = m;
    }
    #pragma unroll
    for (int r = 0; r < 8; ++r){
      float s = 0.f;
      #pragma unroll
      for (int t = 0; t < 6; ++t){ float e = __expf(gacc[t][r] - mx[r]); gacc[t][r] = e; s += e; }
      #pragma unroll
      for (int off = 1; off < 16; off <<= 1) s += __shfl_xor(s, off, 16);
      inv[r] = 1.f / s;
    }
    #pragma unroll
    for (int t = 0; t < 6; ++t){
      float p = 0.f;
      #pragma unroll
      for (int r = 0; r < 8; ++r){
        float sij = gacc[t][r] * inv[r];
        float fv  = (float)fij[(r + 8*c)*96 + t*16 + np];
        p += sij * fv;                      // sum over edges m = r + 8c
      }
      p += __shfl_xor(p, 16, 32);           // combine the two half-waves (m and m+8)
      partial[t] = p;
    }
  }
  if (lane < 16){
    #pragma unroll
    for (int t = 0; t < 6; ++t) sAgg[w*96 + t*16 + lane] = (_Float16)partial[t];
  }
  __syncthreads();

  // ---- out = relu(aggr @ Wg + bg) [+ residual relu(. + x@Ws + bs)] : 16 nodes x DOUT ----
  if (w < DOUT/16){
    const int col = w*16 + np;
    v16h a0 = ldA(sAgg, 96, np, c, 0);
    v16h a1 = ldA(sAgg, 96, np, c, 32);
    v16h a2 = ldA(sAgg, 96, np, c, 64);
    v8f acc = {};
    acc = WMMA_F16(a0, ldB(sWgT, 96, col, c, 0),  acc);
    acc = WMMA_F16(a1, ldB(sWgT, 96, col, c, 32), acc);
    acc = WMMA_F16(a2, ldB(sWgT, 96, col, c, 64), acc);
    float bias = sbg[col];
    v8f res = {};
    if constexpr (RESID){
      v16h r0 = ldA(sXres, 64, np, c, 0);
      v16h r1 = ldA(sXres, 64, np, c, 32);
      res = WMMA_F16(r0, ldB(sWsT, 64, col, c, 0),  res);
      res = WMMA_F16(r1, ldB(sWsT, 64, col, c, 32), res);
    }
    #pragma unroll
    for (int r = 0; r < 8; ++r){
      float h = acc[r] + bias; h = h > 0.f ? h : 0.f;     // conv's own relu
      if constexpr (RESID){
        h = h + res[r] + sbs[col];                        // + x@Ws + bs
        h = h > 0.f ? h : 0.f;                            // final relu
      }
      out[(blockIdx.x*16 + r + 8*c)*DOUT + col] = h;
    }
  }
}

extern "C" void kernel_launch(void* const* d_in, const int* in_sizes, int n_in,
                              void* d_out, int out_size, void* d_ws, size_t ws_size,
                              hipStream_t stream) {
  const float* x    = (const float*)d_in[0];
  const float* pos  = (const float*)d_in[1];
  const int*   idx1 = (const int*)  d_in[2];
  const int*   idx2 = (const int*)  d_in[3];
  const float* Wp1  = (const float*)d_in[4];  const float* bp1 = (const float*)d_in[5];
  const float* Wa1  = (const float*)d_in[6];  const float* ba1 = (const float*)d_in[7];
  const float* Wg1  = (const float*)d_in[8];  const float* bg1 = (const float*)d_in[9];
  const float* Wp2  = (const float*)d_in[10]; const float* bp2 = (const float*)d_in[11];
  const float* Wa2  = (const float*)d_in[12]; const float* ba2 = (const float*)d_in[13];
  const float* Wg2  = (const float*)d_in[14]; const float* bg2 = (const float*)d_in[15];
  const float* Ws   = (const float*)d_in[16]; const float* bs  = (const float*)d_in[17];

  const int N = in_sizes[1] / 3;            // pos is [N,3]
  float* h   = (float*)d_ws;                // intermediate h: N*64 fp32 (16 MB)
  float* outp = (float*)d_out;

  dim3 grid(N / 16), block(512);
  randla_conv<64, 64,  false><<<grid, block, 0, stream>>>(
      x, pos, idx1, Wp1, bp1, Wa1, ba1, Wg1, bg1, nullptr, nullptr, nullptr, h, N);
  randla_conv<64, 128, true ><<<grid, block, 0, stream>>>(
      h, pos, idx2, Wp2, bp2, Wa2, ba2, Wg2, bg2, Ws, bs, x, outp, N);
}
